// TCE_35270271435548
// MI455X (gfx1250) — matrix operand order repaired and checked by reading, not compiled, not run on hardware
//
#include <hip/hip_runtime.h>

typedef __bf16 bf16_t;
typedef __attribute__((ext_vector_type(16))) __bf16 v16bf;
typedef __attribute__((ext_vector_type(8)))  __bf16 v8bf;
typedef __attribute__((ext_vector_type(8)))  float  v8f;

#define B_DIM   512
#define C_DIM   256
#define HWD     384
#define FEA_STRIDE 264                       // bf16 elems per feaT row (padded: 528B row -> conflict-free B loads, 16B aligned)
#define A_STRIDE   264
#define FEA_BYTES  (HWD * FEA_STRIDE * 2)    // 202752
#define A_OFF      FEA_BYTES
#define A_BYTES    (128 * A_STRIDE * 2)      // 67584
#define CADJ_OFF   (A_OFF + A_BYTES)         // 270336
#define LDS_TOTAL  (CADJ_OFF + C_DIM * 4)    // 271360 bytes  (fits 320KB/WGP)

extern __shared__ char smem[];

__global__ __launch_bounds__(256, 1)
void fused_graph_gemm(const float* __restrict__ x,
                      const float* __restrict__ adj,
                      const float* __restrict__ para,
                      float* __restrict__ out)
{
    const int tid   = threadIdx.x;          // 0..255
    const int bid   = blockIdx.x;           // 0..1023
    const int b     = bid >> 1;             // batch
    const int mhalf = bid & 1;              // which 128-channel half of M

    bf16_t* feaT = (bf16_t*)smem;                 // [HWD][FEA_STRIDE]  (N-major -> B K-runs contiguous)
    bf16_t* Amat = (bf16_t*)(smem + A_OFF);       // [128][A_STRIDE]
    float*  sums = (float*)(smem + A_OFF);        // scratch aliased over A region (used before A written)
    float*  cadj = (float*)(smem + CADJ_OFF);     // [256]

    // ---------------- Phase 1: load x[b] -> bf16 transposed LDS + channel sums ----------------
    sums[tid] = 0.0f;
    __syncthreads();

    const float4* xb = (const float4*)(x + (size_t)b * (C_DIM * HWD));
    for (int it = 0; it < 96; ++it) {
        int f4 = it * 256 + tid;             // 0..24575 float4s, fully coalesced
        float4 v = xb[f4];
        int k = f4 / 96;                     // channel row (384 floats = 96 float4 per row)
        int n = (f4 - k * 96) * 4;           // spatial col
        feaT[(n + 0) * FEA_STRIDE + k] = (bf16_t)v.x;
        feaT[(n + 1) * FEA_STRIDE + k] = (bf16_t)v.y;
        feaT[(n + 2) * FEA_STRIDE + k] = (bf16_t)v.z;
        feaT[(n + 3) * FEA_STRIDE + k] = (bf16_t)v.w;
        atomicAdd(&sums[k], v.x + v.y + v.z + v.w);   // ds_add_f32
    }
    __syncthreads();
    cadj[tid] = sums[tid] * (1.0f / (float)HWD);
    __syncthreads();                         // sums fully consumed; A region reusable

    // ---------------- Phase 2: A = adj .* s (s is symmetric: g is even in diff) ----------------
    {
        const float cj = cadj[tid];          // column j = tid
        const int rowbase = mhalf * 128;
        for (int i = 0; i < 128; ++i) {
            const int I = rowbase + i;
            float d   = cadj[I] - cj;        // sign irrelevant (g even)
            float sig = 1.0f / (1.0f + __expf(-d));
            float s   = fabsf(fabsf(sig - 0.5f) - 0.5f) * 2.0f;
            float a   = adj[I * C_DIM + tid] * s;
            Amat[i * A_STRIDE + tid] = (bf16_t)a;
        }
    }
    __syncthreads();

    // ---------------- Phase 3: out[b][mhalf*128 .. +128][:] = A @ fea via bf16 WMMA ----------------
    const int wave = tid >> 5;               // 8 waves
    const int lane = tid & 31;
    const int wm   = wave >> 2;              // 0..1 -> 64-row M block
    const int wn   = wave & 3;               // 0..3 -> 96-col N block
    const int l16  = lane & 15;
    const int lhi  = lane >> 4;              // 0/1 half-wave

    v8f acc[4][6];
    #pragma unroll
    for (int mt = 0; mt < 4; ++mt)
        #pragma unroll
        for (int nt = 0; nt < 6; ++nt) {
            v8f z = {0.f,0.f,0.f,0.f,0.f,0.f,0.f,0.f};
            acc[mt][nt] = z;
        }

    const char* Abase = smem + A_OFF;
    const char* Fbase = smem;

    for (int kt = 0; kt < 8; ++kt) {         // K = 256 in steps of 32
        const int kA = kt * 32 + lhi * 8;    // A lane K-runs: {kA..kA+7, kA+16..kA+23}
        const int kB = kt * 32 + lhi * 8;    // B lane K-runs: {kB..kB+7, kB+16..kB+23}

        v16bf afrag[4], bfrag[6];
        #pragma unroll
        for (int mt = 0; mt < 4; ++mt) {
            const int m = wm * 64 + mt * 16 + l16;                        // local A row
            const v8bf* p = (const v8bf*)(Abase + m * (A_STRIDE * 2) + kA * 2);
            v8bf lo = p[0];                  // ds_load_b128
            v8bf hi = p[2];                  // +32B
            afrag[mt] = __builtin_shufflevector(lo, hi,
                0,1,2,3,4,5,6,7,8,9,10,11,12,13,14,15);
        }
        #pragma unroll
        for (int nt = 0; nt < 6; ++nt) {
            const int n = wn * 96 + nt * 16 + l16;                        // B column (spatial)
            const v8bf* p = (const v8bf*)(Fbase + n * (FEA_STRIDE * 2) + kB * 2);
            v8bf lo = p[0];                  // conflict-free: 528B row stride = 4 banks/lane
            v8bf hi = p[2];
            bfrag[nt] = __builtin_shufflevector(lo, hi,
                0,1,2,3,4,5,6,7,8,9,10,11,12,13,14,15);
        }
        #pragma unroll
        for (int mt = 0; mt < 4; ++mt)
            #pragma unroll
            for (int nt = 0; nt < 6; ++nt)
                acc[mt][nt] = __builtin_amdgcn_wmma_f32_16x16x32_bf16(
                    false, afrag[mt], false, bfrag[nt],
                    (short)0, acc[mt][nt], false, false);
    }

    // ---------------- Epilogue: * para, ReLU, store f32 ----------------
    const size_t outbase = (size_t)b * (C_DIM * HWD);
    #pragma unroll
    for (int mt = 0; mt < 4; ++mt) {
        #pragma unroll
        for (int nt = 0; nt < 6; ++nt) {
            const int n = wn * 96 + nt * 16 + l16;
            #pragma unroll
            for (int r = 0; r < 8; ++r) {
                const int c = mhalf * 128 + wm * 64 + mt * 16 + lhi * 8 + r;  // C/D layout: VGPR r, half-wave adds 8
                float o = acc[mt][nt][r] * para[c * HWD + n];
                out[outbase + (size_t)c * HWD + n] = fmaxf(o, 0.0f);
            }
        }
    }
}

extern "C" void kernel_launch(void* const* d_in, const int* in_sizes, int n_in,
                              void* d_out, int out_size, void* d_ws, size_t ws_size,
                              hipStream_t stream) {
    const float* x    = (const float*)d_in[0];
    const float* adj  = (const float*)d_in[1];
    const float* para = (const float*)d_in[2];
    float* out = (float*)d_out;

    (void)hipFuncSetAttribute((const void*)fused_graph_gemm,
                              hipFuncAttributeMaxDynamicSharedMemorySize, LDS_TOTAL);

    fused_graph_gemm<<<dim3(B_DIM * 2), dim3(256), LDS_TOTAL, stream>>>(x, adj, para, out);
}
